// EKTMSeqModel_61375082660572
// MI455X (gfx1250) — matrix operand order repaired and checked by reading, not compiled, not run on hardware
//
#include <hip/hip_runtime.h>
#include <hip/hip_bf16.h>

#define T_SZ 1024
#define K_SZ 128
#define L_SZ 4096
#define H_SZ 1024

typedef __bf16 v8bf  __attribute__((ext_vector_type(8)));
typedef __bf16 v16bf __attribute__((ext_vector_type(16)));
typedef float  v8f   __attribute__((ext_vector_type(8)));
typedef int    v4i   __attribute__((__vector_size__(16)));   // matches builtin param

typedef __attribute__((address_space(1))) v4i* gv4i_ptr;
typedef __attribute__((address_space(3))) v4i* lv4i_ptr;

#if defined(__has_builtin)
#if __has_builtin(__builtin_amdgcn_global_load_async_to_lds_b128)
#define HAVE_ASYNC_LDS 1
#endif
#endif

// ---------------------------------------------------------------------------
// f32 -> bf16 conversion (4 elements / thread)
// ---------------------------------------------------------------------------
__global__ void k_cvt_bf16(const float* __restrict__ src, __bf16* __restrict__ dst, int n) {
    int i = (blockIdx.x * blockDim.x + threadIdx.x) * 4;
    if (i + 3 < n) {
        float4 v = *(const float4*)(src + i);
        dst[i + 0] = (__bf16)v.x;
        dst[i + 1] = (__bf16)v.y;
        dst[i + 2] = (__bf16)v.z;
        dst[i + 3] = (__bf16)v.w;
    }
}

// ---------------------------------------------------------------------------
// beta = softmax(knowledge_memory @ knowledge_v)   [single block, 1024 thr]
// ---------------------------------------------------------------------------
__global__ void k_beta(const float* __restrict__ km, const float* __restrict__ kv,
                       float* __restrict__ beta) {
    __shared__ float sred[1024];
    __shared__ float skv[K_SZ];
    int tid = threadIdx.x;
    if (tid < K_SZ) skv[tid] = kv[tid];
    __syncthreads();

    float logit[4];
    float lmax = -3.4e38f;
    for (int r = 0; r < 4; ++r) {
        int row = tid + r * 1024;
        const float* p = km + (size_t)row * K_SZ;
        float acc = 0.f;
        for (int k = 0; k < K_SZ; ++k) acc += p[k] * skv[k];
        logit[r] = acc;
        lmax = fmaxf(lmax, acc);
    }
    sred[tid] = lmax;  __syncthreads();
    for (int s = 512; s > 0; s >>= 1) {
        if (tid < s) sred[tid] = fmaxf(sred[tid], sred[tid + s]);
        __syncthreads();
    }
    float gmax = sred[0];  __syncthreads();

    float e[4], lsum = 0.f;
    for (int r = 0; r < 4; ++r) { e[r] = __expf(logit[r] - gmax); lsum += e[r]; }
    sred[tid] = lsum;  __syncthreads();
    for (int s = 512; s > 0; s >>= 1) {
        if (tid < s) sred[tid] += sred[tid + s];
        __syncthreads();
    }
    float inv = 1.f / sred[0];
    for (int r = 0; r < 4; ++r) beta[tid + r * 1024] = e[r] * inv;
}

// ---------------------------------------------------------------------------
// zero s[H]
// ---------------------------------------------------------------------------
__global__ void k_zero(float* __restrict__ p, int n) {
    int i = blockIdx.x * blockDim.x + threadIdx.x;
    if (i < n) p[i] = 0.f;
}

// ---------------------------------------------------------------------------
// s[h] = sum_l beta[l] * h0[l,h]   grid (H/256, L/256), partial + atomicAdd
// ---------------------------------------------------------------------------
__global__ void k_s(const float* __restrict__ beta, const float* __restrict__ h0,
                    float* __restrict__ s) {
    int h = blockIdx.x * 256 + threadIdx.x;
    int l0 = blockIdx.y * 256;
    float acc = 0.f;
    for (int l = l0; l < l0 + 256; ++l)
        acc += beta[l] * h0[(size_t)l * H_SZ + h];
    atomicAdd(&s[h], acc);
}

// ---------------------------------------------------------------------------
// predict_score = [text_v ; s] . score_W + score_b   [single block, 256 thr]
// ---------------------------------------------------------------------------
__global__ void k_score(const float* __restrict__ text_v, const float* __restrict__ s,
                        const float* __restrict__ sw, const float* __restrict__ sb,
                        float* __restrict__ out) {
    __shared__ float sred[256];
    int tid = threadIdx.x;
    float acc = 0.f;
    for (int i = tid; i < T_SZ + H_SZ; i += 256) {
        float pv = (i < T_SZ) ? text_v[i] : s[i - T_SZ];
        acc += pv * sw[i];
    }
    sred[tid] = acc;  __syncthreads();
    for (int r = 128; r > 0; r >>= 1) {
        if (tid < r) sred[tid] += sred[tid + r];
        __syncthreads();
    }
    if (tid == 0) out[0] = sred[0] + sb[0];
}

// ---------------------------------------------------------------------------
// gvec[j] = W_ih[j, off : off+T] . text_v  (off selected by result>=0.5)
// one wave per row, 8 waves / block, grid = 3H/8
// ---------------------------------------------------------------------------
__global__ void k_gvec(const float* __restrict__ W_ih, const float* __restrict__ text_v,
                       const float* __restrict__ result, float* __restrict__ gvec) {
    int tid = threadIdx.x;
    int lane = tid & 31;
    int wave = tid >> 5;
    int row = blockIdx.x * 8 + wave;
    int off = (result[0] >= 0.5f) ? 0 : T_SZ;
    const float* p = W_ih + (size_t)row * (2 * T_SZ) + off;
    float acc = 0.f;
    for (int k = lane; k < T_SZ; k += 32) acc += p[k] * text_v[k];
    for (int m = 16; m > 0; m >>= 1) acc += __shfl_xor(acc, m, 32);
    if (lane == 0) gvec[row] = acc;
}

// ---------------------------------------------------------------------------
// Fused WMMA GEMM (h0 @ W_hh^T, bf16 inputs, f32 acc) + GRU gate epilogue.
// Per block: B slab (3 gates x 16 cols x 1024 K) staged in LDS via async
// copies, shared by all 8 waves. Wave tile: 32(M) x 16(N), all 3 gates.
// Grid: (L/256, H/16) = (16, 64). Dynamic LDS: 48 * 1032 * 2 bytes.
// ---------------------------------------------------------------------------
#define BROW 1032   // padded row stride (bf16 elems): 516 dwords == 4 mod 64 banks

__device__ inline v8f wmma_bf16(v16bf a, v16bf b, v8f c) {
    return __builtin_amdgcn_wmma_f32_16x16x32_bf16(
        false, a, false, b, (short)0, c, false, false);
}

__device__ inline v16bf ld_afrag(const __bf16* p) {
    // per-lane A fragment: K chunk [0..15] relative to p
    v8bf lo = *(const v8bf*)p;
    v8bf hi = *(const v8bf*)(p + 8);
    v16bf r;
    #pragma unroll
    for (int i = 0; i < 8; ++i) { r[i] = lo[i]; r[i + 8] = hi[i]; }
    return r;
}

__device__ inline float fsigmoid(float x) { return 1.f / (1.f + __expf(-x)); }

__global__ void k_gru_wmma(const __bf16* __restrict__ h0b, const __bf16* __restrict__ whhb,
                           const float* __restrict__ h0, const float* __restrict__ beta,
                           const float* __restrict__ gvec,
                           const float* __restrict__ b_ih, const float* __restrict__ b_hh,
                           float* __restrict__ out) {
    extern __shared__ __bf16 Bsh[];   // [48][BROW]

    int tid = threadIdx.x;
    int lane = tid & 31;
    int wave = tid >> 5;
    int m0 = blockIdx.x * 256 + wave * 32;     // 32 rows per wave
    int j0 = blockIdx.y * 16;

    // ---- stage B slab into LDS: 48 rows x 1024 K bf16 (6144 16-byte chunks)
    for (int c = tid; c < 6144; c += 256) {
        int rr = c >> 7;               // 0..47 : gate*16 + col
        int co = (c & 127) << 3;       // K element offset, step 8
        int g  = rr >> 4;
        int r  = rr & 15;
        const __bf16* gsrc = whhb + ((size_t)(g * H_SZ + j0 + r) * H_SZ + co);
        __bf16* ldst = Bsh + rr * BROW + co;
#ifdef HAVE_ASYNC_LDS
        v4i* gsrc_nc = const_cast<v4i*>((const v4i*)gsrc);
        v4i* ldst_v  = (v4i*)ldst;
        __builtin_amdgcn_global_load_async_to_lds_b128(
            (gv4i_ptr)gsrc_nc, (lv4i_ptr)ldst_v, 0, 0);
#else
        *(v8bf*)ldst = *(const v8bf*)gsrc;
#endif
    }
#ifdef HAVE_ASYNC_LDS
#if __has_builtin(__builtin_amdgcn_s_wait_asynccnt)
    __builtin_amdgcn_s_wait_asynccnt(0);
#else
    asm volatile("s_wait_asynccnt 0" ::: "memory");
#endif
#endif
    __syncthreads();

    int halfsel = lane >> 4;                   // 0: lanes 0-15, 1: lanes 16-31
    int kAoff = halfsel * 16;                  // A: K 0..15 vs 16..31
    int kBoff = halfsel * 16;                  // B: K 0..15 vs 16..31
    int arow0 = m0 + (lane & 15);
    int arow1 = arow0 + 16;

    const __bf16* A0 = h0b + (size_t)arow0 * H_SZ;
    const __bf16* A1 = h0b + (size_t)arow1 * H_SZ;
    const __bf16* Bl = Bsh + (size_t)(lane & 15) * BROW + kBoff;  // gate stride 16*BROW

    v8f ar0 = {}, ar1 = {}, az0 = {}, az1 = {}, an0 = {}, an1 = {};

    for (int k0 = 0; k0 < H_SZ; k0 += 32) {
        int kA = k0 + kAoff;
        v16bf a0 = ld_afrag(A0 + kA);
        v16bf a1 = ld_afrag(A1 + kA);
        v16bf br = *(const v16bf*)(Bl + k0);
        ar0 = wmma_bf16(a0, br, ar0);
        ar1 = wmma_bf16(a1, br, ar1);
        v16bf bz = *(const v16bf*)(Bl + 16 * BROW + k0);
        az0 = wmma_bf16(a0, bz, az0);
        az1 = wmma_bf16(a1, bz, az1);
        v16bf bn = *(const v16bf*)(Bl + 32 * BROW + k0);
        an0 = wmma_bf16(a0, bn, an0);
        an1 = wmma_bf16(a1, bn, an1);
    }

    // ---- GRU epilogue (fp32) ----
    int j = j0 + (lane & 15);
    float gv_r = gvec[j],          gv_z = gvec[H_SZ + j],     gv_n = gvec[2 * H_SZ + j];
    float bi_r = b_ih[j],          bi_z = b_ih[H_SZ + j],     bi_n = b_ih[2 * H_SZ + j];
    float bh_r = b_hh[j],          bh_z = b_hh[H_SZ + j],     bh_n = b_hh[2 * H_SZ + j];

    #pragma unroll
    for (int half = 0; half < 2; ++half) {
        v8f vr = half ? ar1 : ar0;
        v8f vz = half ? az1 : az0;
        v8f vn = half ? an1 : an0;
        #pragma unroll
        for (int v = 0; v < 8; ++v) {
            int m = m0 + half * 16 + halfsel * 8 + v;
            float bt = beta[m];
            float r = fsigmoid(bt * gv_r + bi_r + vr[v] + bh_r);
            float z = fsigmoid(bt * gv_z + bi_z + vz[v] + bh_z);
            float n = tanhf(bt * gv_n + bi_n + r * (vn[v] + bh_n));
            float hprev = h0[(size_t)m * H_SZ + j];
            out[1 + (size_t)m * H_SZ + j] = (1.f - z) * n + z * hprev;
        }
    }
}

// ---------------------------------------------------------------------------
// launch
// ---------------------------------------------------------------------------
extern "C" void kernel_launch(void* const* d_in, const int* in_sizes, int n_in,
                              void* d_out, int out_size, void* d_ws, size_t ws_size,
                              hipStream_t stream) {
    const float* text_v  = (const float*)d_in[0];
    const float* know_v  = (const float*)d_in[1];
    const float* result  = (const float*)d_in[2];
    const float* h0      = (const float*)d_in[3];   // [1, L, H]
    const float* km      = (const float*)d_in[4];   // [L, K]
    const float* W_ih    = (const float*)d_in[5];   // [3H, 2T]
    const float* W_hh    = (const float*)d_in[6];   // [3H, H]
    const float* b_ih    = (const float*)d_in[7];
    const float* b_hh    = (const float*)d_in[8];
    const float* score_W = (const float*)d_in[9];
    const float* score_b = (const float*)d_in[10];
    float* out = (float*)d_out;                      // [1 + L*H]

    // workspace layout
    float*  beta = (float*)d_ws;                     // L
    float*  s    = beta + L_SZ;                      // H
    float*  gvec = s + H_SZ;                         // 3H
    __bf16* h0b  = (__bf16*)((char*)d_ws + 64 * 1024);
    __bf16* whhb = h0b + (size_t)L_SZ * H_SZ;

    // 1) bf16 conversions for the big GEMM operands
    {
        int n = L_SZ * H_SZ;
        k_cvt_bf16<<<n / 1024, 256, 0, stream>>>(h0, h0b, n);
        int n2 = 3 * H_SZ * H_SZ;
        k_cvt_bf16<<<n2 / 1024, 256, 0, stream>>>(W_hh, whhb, n2);
    }
    // 2) beta = softmax(km @ know_v)
    k_beta<<<1, 1024, 0, stream>>>(km, know_v, beta);
    // 3) s = beta @ h0
    k_zero<<<H_SZ / 256, 256, 0, stream>>>(s, H_SZ);
    k_s<<<dim3(H_SZ / 256, L_SZ / 256), 256, 0, stream>>>(beta, h0, s);
    // 4) predict_score
    k_score<<<1, 256, 0, stream>>>(text_v, s, score_W, score_b, out);
    // 5) gvec = W_ih[:, half] @ text_v   (rank-1 collapse of x @ W_ih^T)
    k_gvec<<<(3 * H_SZ) / 8, 256, 0, stream>>>(W_ih, text_v, result, gvec);
    // 6) fused WMMA GEMM (LDS-staged B) + GRU gates -> h_new
    size_t lds_bytes = (size_t)48 * BROW * sizeof(__bf16);
    k_gru_wmma<<<dim3(L_SZ / 256, H_SZ / 16), 256, lds_bytes, stream>>>(
        h0b, whhb, h0, beta, gvec, b_ih, b_hh, out);
}
